// Attention_25907242729650
// MI455X (gfx1250) — matrix operand order repaired and checked by reading, not compiled, not run on hardware
//
#include <hip/hip_runtime.h>

// ---------------------------------------------------------------------------
// MHA forward for MI455X (gfx1250, wave32).
//   B=2, S=2048, D=1024, H=16, Dh=64
// Math: V_WMMA_F32_16X16X32_F16 (f16 operands, f32 accumulate).
// Traffic plan (HBM floor ~600 MB @ 23.3 TB/s): attn (537 MB) is written
// exactly once; scores/softmax/ctx are FUSED so attn is never re-read —
// probabilities stay in LDS (128 KB, bank-padded) between the two matmuls.
// ---------------------------------------------------------------------------

#define NB 2
#define NS 2048
#define ND 1024
#define NH 16
#define NDH 64
#define NTOK (NB * NS)          // 4096 tokens
#define ATTN_SCALE 0.125f       // 1/sqrt(64)
#define PITCH 2052              // LDS row pitch in words: 2052 % 64 = 4 -> de-banked

typedef __attribute__((ext_vector_type(16))) _Float16 v16h;
typedef __attribute__((ext_vector_type(8)))  _Float16 v8h;
typedef __attribute__((ext_vector_type(8)))  float    v8f;
typedef __attribute__((ext_vector_type(4)))  float    v4f;

__device__ __forceinline__ v8f wmma_f16(const v16h& a, const v16h& b, const v8f& c) {
    return __builtin_amdgcn_wmma_f32_16x16x32_f16(false, a, false, b, (short)0, c, false, false);
}

// A-fragment (16x32 f16, M=row): lane L<16 -> M=L, K=[k0,k0+16);
// lane>=16 -> M=L-16, K+16. Two contiguous 8-elem runs per lane.
__device__ __forceinline__ v16h load_a_f32(const float* __restrict__ row, int lhi) {
    const float* p = row + lhi * 16;
    v16h a;
#pragma unroll
    for (int i = 0; i < 8; ++i) { a[i] = (_Float16)p[i]; a[i + 8] = (_Float16)p[i + 8]; }
    return a;
}
__device__ __forceinline__ v16h load_a_f16(const _Float16* __restrict__ row, int lhi) {
    const _Float16* p = row + lhi * 16;
    v8h lo = *(const v8h*)p;
    v8h hi = *(const v8h*)(p + 8);
    v16h a;
#pragma unroll
    for (int i = 0; i < 8; ++i) { a[i] = lo[i]; a[i + 8] = hi[i]; }
    return a;
}
// B-fragment (32x16 f16, N=col): lane L<16 -> N=L, K=[k0,k0+16); lane>=16 -> K+16.
__device__ __forceinline__ v16h load_b_f32(const float* __restrict__ col, int lhi) {
    const float* p = col + lhi * 16;
    v8f b0 = *(const v8f*)p;
    v8f b1 = *(const v8f*)(p + 8);
    v16h b;
#pragma unroll
    for (int i = 0; i < 8; ++i) { b[i] = (_Float16)b0[i]; b[i + 8] = (_Float16)b1[i]; }
    return b;
}
__device__ __forceinline__ v16h load_b_f16(const _Float16* __restrict__ col, int lhi) {
    return *(const v16h*)(col + lhi * 16);
}
// B-fragment from LDS f32 row (16B-aligned words, pitch keeps mod-4 alignment).
__device__ __forceinline__ v16h load_b_lds(const float* __restrict__ col, int lhi) {
    const float* p = col + lhi * 16;
    v4f p0 = *(const v4f*)p, p1 = *(const v4f*)(p + 4);
    v4f p2 = *(const v4f*)(p + 8), p3 = *(const v4f*)(p + 12);
    v16h b;
#pragma unroll
    for (int i = 0; i < 4; ++i) {
        b[i] = (_Float16)p0[i]; b[i + 4]  = (_Float16)p1[i];
        b[i + 8] = (_Float16)p2[i]; b[i + 12] = (_Float16)p3[i];
    }
    return b;
}

// ---------------------------------------------------------------------------
// Projection GEMM:  y = x @ W^T + bias   (x: [4096,1024] f32, W: [1024,1024])
// MODE 0: M=feature (A=W), N=token (B=x^T)  -> dst = Qh/Kh [b,h,s,d] f16
// MODE 1: M=token  (A=x),  N=feature (B=W^T)-> dst = Vt [b,h,d,s] f16
// One wave owns a 64(M) x 16(N) strip: B-fragment reused across 4 M-tiles.
// ---------------------------------------------------------------------------
template <int MODE>
__global__ __launch_bounds__(256) void proj_kernel(const float* __restrict__ X,
                                                   const float* __restrict__ W,
                                                   const float* __restrict__ bias,
                                                   _Float16* __restrict__ dst) {
    const int wid  = blockIdx.x * (blockDim.x >> 5) + (threadIdx.x >> 5);
    const int lane = threadIdx.x & 31;
    const int lhi  = lane >> 4;
    const int llo  = lane & 15;

    int m0, n0;
    if (MODE == 0) { m0 = (wid & 15) * 64; n0 = (wid >> 4) * 16; }
    else           { m0 = (wid & 63) * 64; n0 = (wid >> 6) * 16; }

    v8f acc[4];
#pragma unroll
    for (int t = 0; t < 4; ++t)
#pragma unroll
        for (int r = 0; r < 8; ++r) acc[t][r] = 0.0f;

    const float* Amat = (MODE == 0) ? W : X;
    const float* Bmat = (MODE == 0) ? X : W;

    for (int k0 = 0; k0 < ND; k0 += 32) {
        const v16h b = load_b_f32(Bmat + (size_t)(n0 + llo) * ND + k0, lhi);
#pragma unroll
        for (int t = 0; t < 4; ++t) {
            const v16h a = load_a_f32(Amat + (size_t)(m0 + t * 16 + llo) * ND + k0, lhi);
            acc[t] = wmma_f16(a, b, acc[t]);
        }
    }

#pragma unroll
    for (int t = 0; t < 4; ++t) {
        const int mb = m0 + t * 16 + lhi * 8;   // 8 consecutive M per lane
        v8h o;
        size_t base;
        if (MODE == 0) {
            const int tok = n0 + llo, bb = tok >> 11, ss = tok & (NS - 1);
            const int h = mb >> 6, d0 = mb & 63;
#pragma unroll
            for (int r = 0; r < 8; ++r) o[r] = (_Float16)(acc[t][r] + bias[mb + r]);
            base = ((((size_t)bb * NH + h) * NS + ss) * NDH + d0);
        } else {
            const int f = n0 + llo, h = f >> 6, d = f & 63;
            const int bb = mb >> 11, s0 = mb & (NS - 1);
            const float bn = bias[f];
#pragma unroll
            for (int r = 0; r < 8; ++r) o[r] = (_Float16)(acc[t][r] + bn);
            base = ((((size_t)bb * NH + h) * NDH + d) * NS + s0);
        }
        *(v8h*)(dst + base) = o;                 // single b128 store per lane
    }
}

// ---------------------------------------------------------------------------
// FUSED attention core per (b,h, 16-q block), block = 4 waves, 128 KB LDS:
//   phase 1: scores = Qh @ Kh^T into LDS   (M=s A=Kh, N=q B=Qh^T)
//   phase 2: softmax rows in LDS (wave32 shfl_xor), normalized p -> LDS + attn
//   phase 3: ctx = p @ V from LDS          (M=d A=Vt, N=q B=p^T f32->f16)
// attn (537 MB) is written once and never re-read from HBM.
// ---------------------------------------------------------------------------
__global__ __launch_bounds__(128) void attn_fused_kernel(const _Float16* __restrict__ Qh,
                                                         const _Float16* __restrict__ Kh,
                                                         const _Float16* __restrict__ Vt,
                                                         float* __restrict__ attn,
                                                         _Float16* __restrict__ ctx) {
    extern __shared__ float sc[];               // [16][PITCH]
    const int w    = threadIdx.x >> 5;
    const int lane = threadIdx.x & 31;
    const int lhi  = lane >> 4;
    const int llo  = lane & 15;
    const int qblk = blockIdx.x & (NS / 16 - 1);
    const int bh   = blockIdx.x >> 7;

    // ---- phase 1: raw scores ------------------------------------------------
    const int q = qblk * 16 + llo;
    v16h qb[2];
#pragma unroll
    for (int kk = 0; kk < 2; ++kk)
        qb[kk] = load_b_f16(Qh + ((size_t)bh * NS + q) * NDH + kk * 32, lhi);

    for (int st = 0; st < 32; ++st) {           // wave covers s in [w*512, w*512+512)
        const int s0 = w * 512 + st * 16;
        v8f acc;
#pragma unroll
        for (int r = 0; r < 8; ++r) acc[r] = 0.0f;
#pragma unroll
        for (int kk = 0; kk < 2; ++kk) {
            const v16h a = load_a_f16(Kh + ((size_t)bh * NS + s0 + llo) * NDH + kk * 32, lhi);
            acc = wmma_f16(a, qb[kk], acc);
        }
        const int idx = llo * PITCH + s0 + lhi * 8;   // lane col=q, rows=s contiguous
        v4f lo, hi;
#pragma unroll
        for (int r = 0; r < 4; ++r) { lo[r] = acc[r]; hi[r] = acc[r + 4]; }
        *(v4f*)&sc[idx] = lo;
        *(v4f*)&sc[idx + 4] = hi;
    }
    __syncthreads();

    // ---- phase 2: softmax; keep normalized p in LDS, mirror to attn ---------
#pragma unroll
    for (int j = 0; j < 4; ++j) {
        const int qr = w * 4 + j;
        float* row = &sc[(size_t)qr * PITCH];
        float mx = -3.4e38f;
        for (int i = 0; i < 64; ++i) mx = fmaxf(mx, row[i * 32 + lane]);
#pragma unroll
        for (int off = 16; off >= 1; off >>= 1) mx = fmaxf(mx, __shfl_xor(mx, off, 32));
        float sum = 0.0f;
        for (int i = 0; i < 64; ++i) {
            const float p = __expf((row[i * 32 + lane] - mx) * ATTN_SCALE);
            row[i * 32 + lane] = p;
            sum += p;
        }
#pragma unroll
        for (int off = 16; off >= 1; off >>= 1) sum += __shfl_xor(sum, off, 32);
        const float inv = 1.0f / sum;
        float* __restrict__ orow = attn + ((size_t)bh * NS + qblk * 16 + qr) * NS;
        for (int i = 0; i < 64; ++i) {
            const float p = row[i * 32 + lane] * inv;
            row[i * 32 + lane] = p;               // normalized p stays in LDS
            orow[i * 32 + lane] = p;              // sole HBM write of attn
        }
    }
    __syncthreads();

    // ---- phase 3: ctx = p @ V ; wave w owns d-tile [w*16, w*16+16) ----------
    v8f acc;
#pragma unroll
    for (int r = 0; r < 8; ++r) acc[r] = 0.0f;
    const _Float16* __restrict__ vrow = Vt + ((size_t)bh * NDH + w * 16 + llo) * NS;

    for (int k0 = 0; k0 < NS; k0 += 32) {
        __builtin_prefetch(vrow + k0 + 128, 0, 1);          // global_prefetch_b8
        const v16h b = load_b_lds(&sc[(size_t)llo * PITCH + k0], lhi);
        const v16h a = load_a_f16(vrow + k0, lhi);
        acc = wmma_f16(a, b, acc);
    }

    const int bb = bh >> 4, h = bh & 15;
    const int qn = qblk * 16 + llo;               // lane column = q
    v8h o;
#pragma unroll
    for (int r = 0; r < 8; ++r) o[r] = (_Float16)acc[r];
    const size_t base = (((size_t)bb * NS + qn) * ND) + h * NDH + w * 16 + lhi * 8;
    *(v8h*)(ctx + base) = o;                       // b128 store, d contiguous
}

// ---------------------------------------------------------------------------
// Output projection: out = ctx @ Wo^T + bo.  M=feature (A=Wo f32->f16),
// N=token (B=ctx^T f16).  Per-lane 32B f32 stores (contiguous along feature).
// ---------------------------------------------------------------------------
__global__ __launch_bounds__(256) void out_proj_kernel(const _Float16* __restrict__ ctx,
                                                       const float* __restrict__ Wo,
                                                       const float* __restrict__ bo,
                                                       float* __restrict__ out) {
    const int wid  = blockIdx.x * (blockDim.x >> 5) + (threadIdx.x >> 5);
    const int lane = threadIdx.x & 31;
    const int lhi  = lane >> 4;
    const int llo  = lane & 15;
    const int m0   = (wid & 15) * 64;
    const int n0   = (wid >> 4) * 16;

    v8f acc[4];
#pragma unroll
    for (int t = 0; t < 4; ++t)
#pragma unroll
        for (int r = 0; r < 8; ++r) acc[t][r] = 0.0f;

    for (int k0 = 0; k0 < ND; k0 += 32) {
        const v16h b = load_b_f16(ctx + (size_t)(n0 + llo) * ND + k0, lhi);
#pragma unroll
        for (int t = 0; t < 4; ++t) {
            const v16h a = load_a_f32(Wo + (size_t)(m0 + t * 16 + llo) * ND + k0, lhi);
            acc[t] = wmma_f16(a, b, acc[t]);
        }
    }

#pragma unroll
    for (int t = 0; t < 4; ++t) {
        const int mb = m0 + t * 16 + lhi * 8;
        v8f o;
#pragma unroll
        for (int r = 0; r < 8; ++r) o[r] = acc[t][r] + bo[mb + r];
        *(v8f*)(out + (size_t)(n0 + llo) * ND + mb) = o;   // 32B contiguous f32 store
    }
}

// ---------------------------------------------------------------------------
extern "C" void kernel_launch(void* const* d_in, const int* in_sizes, int n_in,
                              void* d_out, int out_size, void* d_ws, size_t ws_size,
                              hipStream_t stream) {
    (void)in_sizes; (void)n_in; (void)out_size; (void)ws_size;
    const float* query  = (const float*)d_in[0];
    const float* key_in = (const float*)d_in[1];
    const float* value  = (const float*)d_in[2];
    const float* Wq = (const float*)d_in[3];  const float* bq = (const float*)d_in[4];
    const float* Wk = (const float*)d_in[5];  const float* bk = (const float*)d_in[6];
    const float* Wv = (const float*)d_in[7];  const float* bv = (const float*)d_in[8];
    const float* Wo = (const float*)d_in[9];  const float* bo = (const float*)d_in[10];

    const size_t NE = (size_t)NTOK * ND;      // 4,194,304 elements per activation
    _Float16* Qh  = (_Float16*)d_ws;          // [b,h,s,d]   f16  8 MB
    _Float16* Kh  = Qh + NE;                  // [b,h,s,d]   f16  8 MB
    _Float16* Vt  = Kh + NE;                  // [b,h,d,s]   f16  8 MB
    _Float16* Ctx = Vt + NE;                  // [t, h*64+d] f16  8 MB

    float* out  = (float*)d_out;              // [B,S,D] f32
    float* attn = out + NE;                   // [B,H,S,S] f32

    // Q/K/V projections (f32 inputs -> f16 WMMA tiles -> f16 activations)
    proj_kernel<0><<<512, 256, 0, stream>>>(query,  Wq, bq, Qh);
    proj_kernel<0><<<512, 256, 0, stream>>>(key_in, Wk, bk, Kh);
    proj_kernel<1><<<512, 256, 0, stream>>>(value,  Wv, bv, Vt);

    // fused scores + softmax + attn-write + ctx;  ~128.3 KB dynamic LDS
    attn_fused_kernel<<<NB * NH * (NS / 16), 128, 16 * PITCH * sizeof(float), stream>>>(
        Qh, Kh, Vt, attn, Ctx);

    // out = ctx @ Wo^T + bo
    out_proj_kernel<<<512, 256, 0, stream>>>(Ctx, Wo, bo, out);
}
